// PersonaGNN_29832842838181
// MI455X (gfx1250) — compile-verified, matrix operand order checked by reading, not run on hardware
//
#include <hip/hip_runtime.h>
#include <hip/hip_bf16.h>

typedef __attribute__((ext_vector_type(16))) _Float16 v16h;
typedef __attribute__((ext_vector_type(8)))  float    v8f;

#define NEG_SLOPE 0.2f

// ---------------------------------------------------------------------------
// GEMM: H[n][o] = sum_k A[n][k] * W[k][o], A: nRows x 128, W: 128 x 128.
// One 256-thread block (8 waves) computes a 128-row x 128-col tile.
// W is converted to f16 and pre-swizzled into LDS in exact B-fragment order
// so each wave reads its v16h B operand as two contiguous ds_load_b128.
// H must be padded to >= gridDim.x*128 rows: stores are UNGUARDED (padded
// rows receive garbage and are never read downstream).
// ---------------------------------------------------------------------------
__global__ __launch_bounds__(256) void gat_gemm_wmma(
    const float* __restrict__ A, const float* __restrict__ W,
    float* __restrict__ H, int nRows)
{
    __shared__ _Float16 lw[128 * 128];   // 32 KB swizzled f16 W image
    __shared__ float    lc[8][16 * 16];  // 8 KB: per-wave C-tile transpose

    const int t = threadIdx.x;

    // --- Fill LDS: W[k][n] -> fragment position (colTile, kBlock, lane, half)
    for (int i = t; i < 128 * 128; i += 256) {
        const int k = i >> 7, n = i & 127;
        const int kb = k >> 5, kl = k & 31;
        const int region = kl >> 4;          // 0 -> halves 0..7, 1 -> halves 8..15
        const int w16 = kl & 15;
        const int g = w16 >> 3;              // lane group (0: lanes 0-15, 1: 16-31)
        const int half = (region << 3) | (w16 & 7);
        const int ct = n >> 4;
        const int lane = (g << 4) | (n & 15);
        lw[((((ct << 2) + kb) << 5) | lane) * 16 + half] = (_Float16)W[i];
    }
    __syncthreads();

    const int wave = t >> 5;
    const int lane = t & 31;
    const int g    = lane >> 4;
    const int l15  = lane & 15;
    const long long rowBase = (long long)blockIdx.x * 128 + wave * 16;
    const long long arow    = rowBase + l15;       // row this lane loads for A
    const bool inb = (arow < (long long)nRows);

    // --- Build the 4 A fragments (K blocks of 32) for this wave's 16 rows.
    v16h afr[4];
    const float* ap = A + arow * 128;
    #pragma unroll
    for (int kb = 0; kb < 4; ++kb) {
        const int k0 = kb * 32 + g * 8;
        float4 f0, f1, f2, f3;
        if (inb) {
            f0 = *(const float4*)(ap + k0);
            f1 = *(const float4*)(ap + k0 + 4);
            f2 = *(const float4*)(ap + k0 + 16);
            f3 = *(const float4*)(ap + k0 + 20);
        } else {
            f0 = f1 = f2 = f3 = make_float4(0.f, 0.f, 0.f, 0.f);
        }
        const float tf[16] = { f0.x,f0.y,f0.z,f0.w, f1.x,f1.y,f1.z,f1.w,
                               f2.x,f2.y,f2.z,f2.w, f3.x,f3.y,f3.z,f3.w };
        v16h a;
        #pragma unroll
        for (int q = 0; q < 16; ++q) a[q] = (_Float16)tf[q];
        afr[kb] = a;
    }

    // Epilogue read mapping: 2 lanes per output row, 8 floats each.
    const int orow  = lane >> 1;          // 0..15
    const int ohalf = (lane & 1) * 8;     // 0 or 8
    float* const obase = H + (rowBase + orow) * 128 + ohalf;

    // --- 8 column tiles x 4 K-blocks of WMMA
    #pragma unroll 1
    for (int ct = 0; ct < 8; ++ct) {
        v8f c = {};
        #pragma unroll
        for (int kb = 0; kb < 4; ++kb) {
            const v16h b = *(const v16h*)&lw[((((ct << 2) + kb) << 5) | lane) * 16];
            c = __builtin_amdgcn_wmma_f32_16x16x32_f16(
                    false, afr[kb], false, b, (short)0, c, false, false);
        }
        // C layout: lane group g covers M = g*8 + r, column = lane & 15.
        // Stage through wave-private LDS tile, then store row-major b128 pairs.
        #pragma unroll
        for (int r = 0; r < 8; ++r)
            lc[wave][(g * 8 + r) * 16 + l15] = c[r];
        // LDS is in-order within a wave; compiler inserts s_wait_dscnt.
        const float4 o0 = *(const float4*)&lc[wave][orow * 16 + ohalf];
        const float4 o1 = *(const float4*)&lc[wave][orow * 16 + ohalf + 4];
        *(float4*)(obase + ct * 16)     = o0;
        *(float4*)(obase + ct * 16 + 4) = o1;
    }
}

// ---------------------------------------------------------------------------
// Per-node attention logits: as[n] = h[n].a_src ; ad[n] = h[n].a_dst
// One wave (32 lanes x float4) per node.
// ---------------------------------------------------------------------------
__global__ __launch_bounds__(256) void gat_alphas(
    const float* __restrict__ H, const float* __restrict__ a_s,
    const float* __restrict__ a_d, float* __restrict__ as,
    float* __restrict__ ad, int n)
{
    const int wid  = (int)((blockIdx.x * 256u + threadIdx.x) >> 5);
    const int lane = threadIdx.x & 31;
    if (wid >= n) return;
    const float4 h4 = *(const float4*)(H + (long long)wid * 128 + lane * 4);
    const float4 s4 = *(const float4*)(a_s + lane * 4);
    const float4 d4 = *(const float4*)(a_d + lane * 4);
    float s = h4.x * s4.x + h4.y * s4.y + h4.z * s4.z + h4.w * s4.w;
    float d = h4.x * d4.x + h4.y * d4.y + h4.z * d4.z + h4.w * d4.w;
    #pragma unroll
    for (int off = 16; off > 0; off >>= 1) {
        s += __shfl_xor(s, off, 32);
        d += __shfl_xor(d, off, 32);
    }
    if (lane == 0) { as[wid] = s; ad[wid] = d; }
}

// Order-preserving float <-> uint map for atomicMax on floats.
__device__ __forceinline__ unsigned f2ord(float f) {
    unsigned u = __float_as_uint(f);
    return (u >> 31) ? ~u : (u | 0x80000000u);
}
__device__ __forceinline__ float ord2f(unsigned u) {
    return (u >> 31) ? __uint_as_float(u & 0x7fffffffu) : __uint_as_float(~u);
}

// e = leaky_relu(as[src] + ad[dst]); segment max into m[dst]
__global__ __launch_bounds__(256) void gat_edge_max(
    const int* __restrict__ ei, int E, int TE,
    const float* __restrict__ as, const float* __restrict__ ad,
    float* __restrict__ ebuf, unsigned* __restrict__ mord)
{
    const int e = blockIdx.x * 256 + threadIdx.x;
    if (e >= TE) return;
    int src, dst;
    if (e < E) { src = ei[e]; dst = ei[E + e]; } else { src = dst = e - E; }
    float v = as[src] + ad[dst];
    v = (v > 0.f) ? v : NEG_SLOPE * v;
    ebuf[e] = v;
    atomicMax(&mord[dst], f2ord(v));
}

// w = exp(e - m[dst]); denom[dst] += w (in-place overwrite ebuf with w)
__global__ __launch_bounds__(256) void gat_edge_exp(
    const int* __restrict__ ei, int E, int TE,
    const unsigned* __restrict__ mord, float* __restrict__ ebuf,
    float* __restrict__ denom)
{
    const int e = blockIdx.x * 256 + threadIdx.x;
    if (e >= TE) return;
    const int dst = (e < E) ? ei[E + e] : (e - E);
    const float w = expf(ebuf[e] - ord2f(mord[dst]));
    ebuf[e] = w;
    atomicAdd(&denom[dst], w);
}

// agg[dst] += h[src] * (w / denom[dst]) -- one wave per edge, float4 per lane
__global__ __launch_bounds__(256) void gat_edge_agg(
    const int* __restrict__ ei, int E, int TE,
    const float* __restrict__ H, const float* __restrict__ ebuf,
    const float* __restrict__ denom, float* __restrict__ agg)
{
    const int gw   = (int)((blockIdx.x * 256u + threadIdx.x) >> 5);
    const int lane = threadIdx.x & 31;
    if (gw >= TE) return;
    int src, dst;
    if (gw < E) { src = ei[gw]; dst = ei[E + gw]; } else { src = dst = gw - E; }
    const float a = ebuf[gw] / denom[dst];
    const float4 h4 = *(const float4*)(H + (long long)src * 128 + lane * 4);
    float* o = agg + (long long)dst * 128 + lane * 4;
    atomicAdd(o + 0, h4.x * a);
    atomicAdd(o + 1, h4.y * a);
    atomicAdd(o + 2, h4.z * a);
    atomicAdd(o + 3, h4.w * a);
}

// g = relu(g + b)   (layer-1 activation, elementwise over N*128)
__global__ __launch_bounds__(256) void gat_bias_relu(
    float* __restrict__ g, const float* __restrict__ b, long long total)
{
    const long long i = (long long)blockIdx.x * 256 + threadIdx.x;
    if (i >= total) return;
    const float v = g[i] + b[(int)(i & 127)];
    g[i] = (v > 0.f) ? v : 0.f;
}

// Column partial sums over 256-row stripes, atomics into outSum[128].
__global__ __launch_bounds__(128) void gat_colsum(
    const float* __restrict__ agg, int n, float* __restrict__ outSum)
{
    const int j = threadIdx.x;
    const long long r0 = (long long)blockIdx.x * 256;
    long long r1 = r0 + 256; if (r1 > n) r1 = n;
    float s = 0.f;
    for (long long r = r0; r < r1; ++r) s += agg[r * 128 + j];
    atomicAdd(&outSum[j], s);
}

__global__ __launch_bounds__(128) void gat_finalize(
    const float* __restrict__ outSum, const float* __restrict__ b2,
    float* __restrict__ out, float invN)
{
    const int j = threadIdx.x;
    out[j] = outSum[j] * invN + b2[j];
}

// ---------------------------------------------------------------------------
extern "C" void kernel_launch(void* const* d_in, const int* in_sizes, int n_in,
                              void* d_out, int out_size, void* d_ws, size_t ws_size,
                              hipStream_t stream)
{
    const float* x      = (const float*)d_in[0];
    const int*   ei     = (const int*)  d_in[1];
    const float* W1     = (const float*)d_in[2];
    const float* a_src1 = (const float*)d_in[3];
    const float* a_dst1 = (const float*)d_in[4];
    const float* b1     = (const float*)d_in[5];
    const float* W2     = (const float*)d_in[6];
    const float* a_src2 = (const float*)d_in[7];
    const float* a_dst2 = (const float*)d_in[8];
    const float* b2     = (const float*)d_in[9];
    float* out = (float*)d_out;

    const int N  = in_sizes[0] / 128;
    const int E  = in_sizes[1] / 2;
    const int TE = E + N;                  // + self loops
    const int NP = (N + 127) & ~127;       // rows padded to GEMM tile multiple

    // Workspace carve-out (H and G padded: GEMM stores are unguarded)
    float* ws = (float*)d_ws;
    float*    H      = ws;                 ws += (size_t)NP * 128;
    float*    G      = ws;                 ws += (size_t)NP * 128;
    float*    as     = ws;                 ws += N;
    float*    ad     = ws;                 ws += N;
    unsigned* mord   = (unsigned*)ws;      ws += N;
    float*    denom  = ws;                 ws += N;
    float*    ebuf   = ws;                 ws += TE;
    float*    outSum = ws;                 ws += 128;

    const unsigned gGemm  = (unsigned)(NP / 128);
    const unsigned gNode  = (unsigned)((N + 7) / 8);          // 8 waves/block
    const unsigned gEdgeT = (unsigned)((TE + 255) / 256);     // thread/edge
    const unsigned gEdgeW = (unsigned)((TE + 7) / 8);         // wave/edge
    const unsigned gElem  = (unsigned)(((long long)N * 128 + 255) / 256);
    const unsigned gCol   = (unsigned)((N + 255) / 256);

    // ---------------- Layer 1 ----------------
    hipMemsetAsync(G,     0, (size_t)N * 128 * sizeof(float), stream);
    hipMemsetAsync(mord,  0, (size_t)N * sizeof(unsigned),    stream);
    hipMemsetAsync(denom, 0, (size_t)N * sizeof(float),       stream);

    gat_gemm_wmma<<<gGemm, 256, 0, stream>>>(x, W1, H, N);
    gat_alphas  <<<gNode, 256, 0, stream>>>(H, a_src1, a_dst1, as, ad, N);
    gat_edge_max<<<gEdgeT, 256, 0, stream>>>(ei, E, TE, as, ad, ebuf, mord);
    gat_edge_exp<<<gEdgeT, 256, 0, stream>>>(ei, E, TE, mord, ebuf, denom);
    gat_edge_agg<<<gEdgeW, 256, 0, stream>>>(ei, E, TE, H, ebuf, denom, G);
    gat_bias_relu<<<gElem, 256, 0, stream>>>(G, b1, (long long)N * 128);

    // ---------------- Layer 2 ----------------
    gat_gemm_wmma<<<gGemm, 256, 0, stream>>>(G, W2, H, N);
    hipMemsetAsync(G,     0, (size_t)N * 128 * sizeof(float), stream);
    hipMemsetAsync(mord,  0, (size_t)N * sizeof(unsigned),    stream);
    hipMemsetAsync(denom, 0, (size_t)N * sizeof(float),       stream);

    gat_alphas  <<<gNode, 256, 0, stream>>>(H, a_src2, a_dst2, as, ad, N);
    gat_edge_max<<<gEdgeT, 256, 0, stream>>>(ei, E, TE, as, ad, ebuf, mord);
    gat_edge_exp<<<gEdgeT, 256, 0, stream>>>(ei, E, TE, mord, ebuf, denom);
    gat_edge_agg<<<gEdgeW, 256, 0, stream>>>(ei, E, TE, H, ebuf, denom, G);

    // ---------------- Mean over nodes + bias ----------------
    hipMemsetAsync(outSum, 0, 128 * sizeof(float), stream);
    gat_colsum  <<<gCol, 128, 0, stream>>>(G, N, outSum);
    gat_finalize<<<1, 128, 0, stream>>>(outSum, b2, out, 1.0f / (float)N);
}
